// DiffAttention_16973710754153
// MI455X (gfx1250) — matrix operand order repaired and checked by reading, not compiled, not run on hardware
//
#include <hip/hip_runtime.h>

#define DIMX   768
#define NSEQ   2048
#define BATCH  4
#define HEADS  12
#define EHEADS 6
#define HD     64
#define HD2    128

typedef __attribute__((ext_vector_type(16))) __bf16       v16bf;
typedef __attribute__((ext_vector_type(8)))  float        v8f;
typedef __attribute__((ext_vector_type(4)))  unsigned int v4u;
typedef __attribute__((ext_vector_type(4)))  unsigned int u32x4;
typedef __attribute__((ext_vector_type(8)))  int          i32x8;
typedef __attribute__((ext_vector_type(4)))  int          i32x4;

// ---------------- helpers ----------------

static __device__ __forceinline__ __bf16 f2bf(float f) {
  union { float f; unsigned int u; } a; a.f = f;
  unsigned int u = a.u;
  unsigned int r = (u + 0x7FFFu + ((u >> 16) & 1u)) >> 16;  // RNE
  union { unsigned short s; __bf16 b; } c; c.s = (unsigned short)r;
  return c.b;
}

static __device__ __forceinline__ v8f zero8() {
  v8f z;
#pragma unroll
  for (int i = 0; i < 8; ++i) z[i] = 0.0f;
  return z;
}

static __device__ __forceinline__ v8f wmma_bf16(v16bf a, v16bf b, v8f c) {
  // D = A(16x32 bf16) * B(32x16 bf16) + C(16x16 f32)
  return __builtin_amdgcn_wmma_f32_16x16x32_bf16(
      /*neg_a=*/false, a, /*neg_b=*/false, b,
      /*c_mod=*/(short)0, c, /*reuse_a=*/false, /*reuse_b=*/false);
}

// A fragment: 16 rows x 32 K, row-major source (pitch in elements).
// ISA layout: lane m=l&15, half=l>>4; elems 0..7 -> K = k0+half*8+j,
// elems 8..15 -> K = k0+16+half*8+(j-8).
static __device__ __forceinline__ v16bf frag_a(const __bf16* row0, int pitch,
                                               int k0, int lane) {
  int m = lane & 15, half = lane >> 4;
  const __bf16* p = row0 + (size_t)m * pitch + k0;
  union { v16bf v; v4u q[2]; } r;
  r.q[0] = *(const v4u*)(p + half * 8);
  r.q[1] = *(const v4u*)(p + 16 + half * 8);
  return r.v;
}

// B fragment: 32 K x 16 cols; source holds one output-column per row with K
// contiguous (i.e. B^T row-major). Lane col n=l&15, elems 0..7 -> K = k0+half*8+j,
// elems 8..15 -> K = k0+16+half*8+(j-8).
static __device__ __forceinline__ v16bf frag_b(const __bf16* col0, int pitch,
                                               int k0, int lane) {
  int n = lane & 15, half = lane >> 4;
  const __bf16* p = col0 + (size_t)n * pitch + k0;
  union { v16bf v; v4u q[2]; } r;
  r.q[0] = *(const v4u*)(p + half * 8);
  r.q[1] = *(const v4u*)(p + 16 + half * 8);
  return r.v;
}

// ---------------- TDM tile load: 64 rows x 32 bf16, src stride 768 elems,
// ---------------- dst LDS packed with 8-element pad per row (pitch 40).

#if __has_builtin(__builtin_amdgcn_tensor_load_to_lds)
#define USE_TDM 1
#else
#define USE_TDM 0
#endif

#if USE_TDM
static __device__ __forceinline__ void tdm_load_tile_64x32(const __bf16* gsrc,
                                                           void* lds_dst) {
  unsigned long long ga = (unsigned long long)(size_t)gsrc;
  unsigned lds_off = (unsigned)(size_t)lds_dst;  // addr[31:0] = LDS offset

  u32x4 g0;
  g0[0] = 1u;                                     // count=1, user mode
  g0[1] = lds_off;                                // lds_addr
  g0[2] = (unsigned)(ga & 0xFFFFFFFFu);           // global_addr[31:0]
  g0[3] = (unsigned)((ga >> 32) & 0x01FFFFFFu)    // global_addr[56:32]
          | (2u << 30);                           // type=2 ("image")

  i32x8 g1;
  // data_size=1 (2B), pad_enable, pad_interval code 3 (16 DW = 64B = row),
  // pad_amount code 3 (4 DW = 16B = 8 elems)  -> LDS pitch 40 elements
  g1[0] = (1 << 16) | (1 << 20) | (3 << 22) | (3 << 25);
  g1[1] = (DIMX & 0xFFFF) << 16;                  // tensor_dim0[15:0] = 768
  g1[2] = ((DIMX >> 16) & 0xFFFF)                 // tensor_dim0[31:16]
          | ((8192 & 0xFFFF) << 16);              // tensor_dim1[15:0]
  g1[3] = ((8192 >> 16) & 0xFFFF)                 // tensor_dim1[31:16]
          | (32 << 16);                           // tile_dim0 = 32
  g1[4] = 64;                                     // tile_dim1=64, tile_dim2=0
  g1[5] = DIMX;                                   // tensor_dim0_stride lo32
  g1[6] = 0;                                      // stride0 hi16, stride1 lo16
  g1[7] = 0;                                      // stride1 hi32

  i32x4 gz;
#pragma unroll
  for (int i = 0; i < 4; ++i) gz[i] = 0;

#if __clang_major__ >= 23
  i32x8 gz8;
#pragma unroll
  for (int i = 0; i < 8; ++i) gz8[i] = 0;
  __builtin_amdgcn_tensor_load_to_lds(g0, g1, gz, gz, gz8, 0);
#else
  __builtin_amdgcn_tensor_load_to_lds(g0, g1, gz, gz, 0);
#endif
}
#endif  // USE_TDM

// ---------------- conversion kernels ----------------

__global__ void cvt_kernel(const float* __restrict__ src,
                           __bf16* __restrict__ dst, int n) {
  int i = blockIdx.x * blockDim.x + threadIdx.x;
  if (i < n) dst[i] = f2bf(src[i]);
}

// dst[n*768+k] = src[k*768+n]  (pre-transpose weights so GEMM B-frags are contiguous)
__global__ void cvt_t_kernel(const float* __restrict__ src,
                             __bf16* __restrict__ dst) {
  int i = blockIdx.x * blockDim.x + threadIdx.x;
  if (i >= DIMX * DIMX) return;
  int n = i / DIMX, k = i - n * DIMX;
  dst[i] = f2bf(src[k * DIMX + n]);
}

// ---------------- GEMM: C(8192x768) = A(8192x768) @ B + bias ----------------
// mode 0: q -> (B,12,N,64) bf16, value scaled by 0.125 after bias
// mode 1: k -> (B,12,N,64) bf16
// mode 2: v -> transposed (B,6,128,N) bf16
// mode 3: plain (row,col) f32 -> d_out

__global__ __launch_bounds__(128) void gemm_kernel(
    const __bf16* __restrict__ A, const __bf16* __restrict__ Bt,
    const float* __restrict__ bias, void* __restrict__ outp,
    int mode, float scale) {
  __shared__ __bf16 As[64][40];
  __shared__ __bf16 Bs[64][40];

  int tid  = threadIdx.x;
  int lane = tid & 31;
  int wave = tid >> 5;
  int m0 = blockIdx.x * 64;
  int n0 = blockIdx.y * 64;

  v8f acc[4];
#pragma unroll
  for (int i = 0; i < 4; ++i) acc[i] = zero8();

  int lrow = tid >> 1;
  int lk   = (tid & 1) * 16;
  (void)lrow; (void)lk;

  for (int k0 = 0; k0 < DIMX; k0 += 32) {
#if USE_TDM
    // One wave drives the Tensor Data Mover for both tiles; TENSORcnt tracks
    // completion, then the workgroup barrier publishes the LDS data.
    if (wave == 0) {
      tdm_load_tile_64x32(A + (size_t)m0 * DIMX + k0, &As[0][0]);
      tdm_load_tile_64x32(Bt + (size_t)n0 * DIMX + k0, &Bs[0][0]);
      __builtin_amdgcn_s_wait_tensorcnt(0);
    }
#else
    {
      const __bf16* sa = A + (size_t)(m0 + lrow) * DIMX + k0 + lk;
      *(v4u*)(&As[lrow][lk])     = *(const v4u*)(sa);
      *(v4u*)(&As[lrow][lk + 8]) = *(const v4u*)(sa + 8);
      const __bf16* sb = Bt + (size_t)(n0 + lrow) * DIMX + k0 + lk;
      *(v4u*)(&Bs[lrow][lk])     = *(const v4u*)(sb);
      *(v4u*)(&Bs[lrow][lk + 8]) = *(const v4u*)(sb + 8);
    }
#endif
    if (k0 + 32 < DIMX) {  // global_prefetch_b8 for next tiles
      __builtin_prefetch(A + (size_t)(m0 + (tid >> 1)) * DIMX + k0 + 32, 0, 1);
      __builtin_prefetch(Bt + (size_t)(n0 + (tid >> 1)) * DIMX + k0 + 32, 0, 1);
    }
    __syncthreads();

    // preload all fragments, then issue WMMAs back-to-back for XDL ILP
    v16bf af = frag_a(&As[wave * 16][0], 40, 0, lane);
    v16bf bfr[4];
#pragma unroll
    for (int nt = 0; nt < 4; ++nt) bfr[nt] = frag_b(&Bs[nt * 16][0], 40, 0, lane);
#pragma unroll
    for (int nt = 0; nt < 4; ++nt) acc[nt] = wmma_bf16(af, bfr[nt], acc[nt]);
    __syncthreads();
  }

  // store C tiles: lane (half,nl); VGPR r -> row = half*8 + r
  int half = lane >> 4, nl = lane & 15;
#pragma unroll
  for (int nt = 0; nt < 4; ++nt) {
    int col = n0 + nt * 16 + nl;
    float bv = bias[col];
#pragma unroll
    for (int r = 0; r < 8; ++r) {
      int row = m0 + wave * 16 + half * 8 + r;
      float v = (acc[nt][r] + bv) * scale;
      if (mode == 3) {
        ((float*)outp)[(size_t)row * DIMX + col] = v;
      } else {
        int b = row >> 11, n = row & (NSEQ - 1);
        size_t idx;
        if (mode == 2) {
          int h = col >> 7, d = col & 127;
          idx = (((size_t)(b * EHEADS + h) * HD2 + d) * NSEQ + n);
        } else {
          int h = col >> 6, d = col & 63;
          idx = (((size_t)(b * HEADS + h) * NSEQ + n) * HD + d);
        }
        ((__bf16*)outp)[idx] = f2bf(v);
      }
    }
  }
}

// ---------------- fused differential flash attention ----------------
// grid: (N/64, EHEADS, BATCH); block: 128 threads (4 waves, 16 queries/wave)

__global__ __launch_bounds__(128) void diffattn_kernel(
    const __bf16* __restrict__ qw, const __bf16* __restrict__ kw,
    const __bf16* __restrict__ vT,
    const float* __restrict__ lq1, const float* __restrict__ lk1,
    const float* __restrict__ lq2, const float* __restrict__ lk2,
    const float* __restrict__ nw, __bf16* __restrict__ ao) {
  __shared__ float  Sld[64][68];
  __shared__ __bf16 Pld[64][72];
  __shared__ float  Old[64][132];
  __shared__ float  Mrow[2][64];
  __shared__ float  Lrow[2][64];
  __shared__ float  Arow[64];
  __shared__ float  s_lam;

  int tid  = threadIdx.x;
  int lane = tid & 31;
  int wave = tid >> 5;
  int half = lane >> 4, nl = lane & 15;
  int q0 = blockIdx.x * 64;
  int h  = blockIdx.y;
  int b  = blockIdx.z;

  if (tid == 0) {
    float s1 = 0.f, s2 = 0.f;
    for (int i = 0; i < HD; ++i) { s1 += lq1[i] * lk1[i]; s2 += lq2[i] * lk2[i]; }
    s_lam = __expf(s1) - __expf(s2) + 0.8f;
  }
  if (tid < 64) {
    Mrow[0][tid] = -1e30f; Mrow[1][tid] = -1e30f;
    Lrow[0][tid] = 0.f;    Lrow[1][tid] = 0.f;
  }
  __syncthreads();

  // persistent Q fragments (already scaled by dh^-0.5 in the GEMM)
  v16bf qf[2][2];
#pragma unroll
  for (int c = 0; c < 2; ++c) {
    const __bf16* qb =
        qw + ((size_t)(b * HEADS + 2 * h + c) * NSEQ + q0 + wave * 16) * HD;
#pragma unroll
    for (int kk = 0; kk < 2; ++kk) qf[c][kk] = frag_a(qb, HD, kk * 32, lane);
  }

  v8f oacc[2][8];
#pragma unroll
  for (int c = 0; c < 2; ++c)
#pragma unroll
    for (int dt = 0; dt < 8; ++dt) oacc[c][dt] = zero8();

  const __bf16* vbase = vT + (size_t)(b * EHEADS + h) * HD2 * NSEQ;

  for (int kt = 0; kt < NSEQ; kt += 64) {
    for (int c = 0; c < 2; ++c) {
      // ---- scores: 16q x 64k per wave ----
      const __bf16* kb =
          kw + ((size_t)(b * HEADS + 2 * h + c) * NSEQ + kt) * HD;
      v8f s[4];
#pragma unroll
      for (int nt = 0; nt < 4; ++nt) s[nt] = zero8();
      {
        v16bf kf[4][2];
#pragma unroll
        for (int nt = 0; nt < 4; ++nt)
#pragma unroll
          for (int kk = 0; kk < 2; ++kk)
            kf[nt][kk] = frag_b(kb + (size_t)nt * 16 * HD, HD, kk * 32, lane);
#pragma unroll
        for (int nt = 0; nt < 4; ++nt)
#pragma unroll
          for (int kk = 0; kk < 2; ++kk)
            s[nt] = wmma_bf16(qf[c][kk], kf[nt][kk], s[nt]);
      }
#pragma unroll
      for (int nt = 0; nt < 4; ++nt)
#pragma unroll
        for (int r = 0; r < 8; ++r)
          Sld[wave * 16 + half * 8 + r][nt * 16 + nl] = s[nt][r];
      __syncthreads();

      // ---- online softmax (one row per thread) ----
      if (tid < 64) {
        float mold = Mrow[c][tid];
        float mx = mold;
        for (int j = 0; j < 64; ++j) mx = fmaxf(mx, Sld[tid][j]);
        float alpha = __expf(mold - mx);
        float sum = 0.f;
        for (int j = 0; j < 64; ++j) {
          float e = __expf(Sld[tid][j] - mx);
          Pld[tid][j] = f2bf(e);
          sum += e;
        }
        Lrow[c][tid] = Lrow[c][tid] * alpha + sum;
        Mrow[c][tid] = mx;
        Arow[tid] = alpha;
      }
      __syncthreads();

      // ---- rescale accumulators, then P @ V^T ----
      float alr[8];
#pragma unroll
      for (int r = 0; r < 8; ++r) alr[r] = Arow[wave * 16 + half * 8 + r];
      v16bf pf[2];
#pragma unroll
      for (int kk = 0; kk < 2; ++kk)
        pf[kk] = frag_a(&Pld[wave * 16][0], 72, kk * 32, lane);
#pragma unroll
      for (int dt = 0; dt < 8; ++dt) {
#pragma unroll
        for (int r = 0; r < 8; ++r) oacc[c][dt][r] *= alr[r];
        v16bf vf0 = frag_b(vbase + (size_t)dt * 16 * NSEQ + kt, NSEQ, 0, lane);
        v16bf vf1 = frag_b(vbase + (size_t)dt * 16 * NSEQ + kt, NSEQ, 32, lane);
        oacc[c][dt] = wmma_bf16(pf[0], vf0, oacc[c][dt]);
        oacc[c][dt] = wmma_bf16(pf[1], vf1, oacc[c][dt]);
      }
      __syncthreads();
    }
  }

  // ---- combine components, normalize rows, RMSNorm, emit bf16 ----
  float lam = s_lam;
#pragma unroll
  for (int dt = 0; dt < 8; ++dt)
#pragma unroll
    for (int r = 0; r < 8; ++r) {
      int row = wave * 16 + half * 8 + r;
      float o = oacc[0][dt][r] / Lrow[0][row] -
                lam * (oacc[1][dt][r] / Lrow[1][row]);
      Old[row][dt * 16 + nl] = o;
    }
  __syncthreads();

  if (tid < 64) {
    float ss = 0.f;
    for (int d = 0; d < HD2; ++d) ss += Old[tid][d] * Old[tid][d];
    float rms = rsqrtf(ss * (1.0f / HD2) + 1e-5f);
    size_t orow = ((size_t)b * NSEQ + q0 + tid) * DIMX + (size_t)h * HD2;
    for (int d = 0; d < HD2; ++d)
      ao[orow + d] = f2bf(Old[tid][d] * rms * nw[d] * 0.2f);
  }
}

// ---------------- host launch ----------------

extern "C" void kernel_launch(void* const* d_in, const int* in_sizes, int n_in,
                              void* d_out, int out_size, void* d_ws,
                              size_t ws_size, hipStream_t stream) {
  (void)in_sizes; (void)n_in; (void)out_size; (void)ws_size;
  const float* x   = (const float*)d_in[0];
  const float* Wq  = (const float*)d_in[1];
  const float* bq  = (const float*)d_in[2];
  const float* Wk  = (const float*)d_in[3];
  const float* bk  = (const float*)d_in[4];
  const float* Wv  = (const float*)d_in[5];
  const float* bv  = (const float*)d_in[6];
  const float* Wo  = (const float*)d_in[7];
  const float* bo  = (const float*)d_in[8];
  const float* lq1 = (const float*)d_in[9];
  const float* lk1 = (const float*)d_in[10];
  const float* lq2 = (const float*)d_in[11];
  const float* lk2 = (const float*)d_in[12];
  const float* nw  = (const float*)d_in[13];

  const size_t NE = (size_t)BATCH * NSEQ * DIMX;  // 6,291,456
  char* ws = (char*)d_ws;
  __bf16* xbf = (__bf16*)(ws);
  __bf16* qw  = (__bf16*)(ws + NE * 2);
  __bf16* kw  = (__bf16*)(ws + NE * 4);
  __bf16* vT  = (__bf16*)(ws + NE * 6);
  __bf16* ao  = (__bf16*)(ws + NE * 8);
  __bf16* WqT = (__bf16*)(ws + NE * 10);
  __bf16* WkT = WqT + DIMX * DIMX;
  __bf16* WvT = WkT + DIMX * DIMX;
  __bf16* WoT = WvT + DIMX * DIMX;

  cvt_kernel<<<(int)((NE + 255) / 256), 256, 0, stream>>>(x, xbf, (int)NE);
  cvt_t_kernel<<<(DIMX * DIMX + 255) / 256, 256, 0, stream>>>(Wq, WqT);
  cvt_t_kernel<<<(DIMX * DIMX + 255) / 256, 256, 0, stream>>>(Wk, WkT);
  cvt_t_kernel<<<(DIMX * DIMX + 255) / 256, 256, 0, stream>>>(Wv, WvT);
  cvt_t_kernel<<<(DIMX * DIMX + 255) / 256, 256, 0, stream>>>(Wo, WoT);

  dim3 gg(BATCH * NSEQ / 64, DIMX / 64);
  gemm_kernel<<<gg, 128, 0, stream>>>(xbf, WqT, bq, (void*)qw, 0, 0.125f);
  gemm_kernel<<<gg, 128, 0, stream>>>(xbf, WkT, bk, (void*)kw, 1, 1.0f);
  gemm_kernel<<<gg, 128, 0, stream>>>(xbf, WvT, bv, (void*)vT, 2, 1.0f);

  diffattn_kernel<<<dim3(NSEQ / 64, EHEADS, BATCH), 128, 0, stream>>>(
      qw, kw, vT, lq1, lk1, lq2, lk2, nw, ao);

  gemm_kernel<<<gg, 128, 0, stream>>>(ao, WoT, bo, d_out, 3, 1.0f);
}
